// VectorQuantization_7000796692958
// MI455X (gfx1250) — compile-verified, matrix operand order checked by reading, not run on hardware
//
#include <hip/hip_runtime.h>

// ---------------------------------------------------------------------------
// VQ codebook search on gfx1250 via bf16 WMMA.
//   x:      [B=16, D=512, N=8192] f32   (columns along N are the vectors)
//   embed:  [K=2048, D=512] f32
// Outputs (concat in d_out, float*):
//   quantize [B, D, N] f32, then embed_ind [B, N] (index values as float)
//
// GEMM kernel: 8 waves/block, one 16-row M-tile per wave; A operand (16x512
// bf16) register-resident for the whole K loop. B fragments (pre-swizzled to
// WMMA register images) are staged per code-tile into a double-buffered LDS
// ring shared by all 8 waves, prefetched one tile ahead (async-to-LDS when
// the toolchain exposes the builtin).
// ---------------------------------------------------------------------------

typedef __attribute__((ext_vector_type(16))) __bf16 v16bf;
typedef __attribute__((ext_vector_type(8)))  float  v8f;
typedef __attribute__((ext_vector_type(8)))  int    v8i;
typedef __attribute__((__vector_size__(16))) int    vsi4;   // for async-LDS builtin

#define B_  16
#define D_  512
#define N_  8192
#define K_  2048
#define NTILES_K 128                       // code tiles of 16
#define NCHUNK_D 16                        // reduction chunks of 32
#define FRAG_DW  256                       // dwords per 32x16 bf16 fragment
#define KT_DW    (NCHUNK_D * FRAG_DW)      // 4096 dwords = 16 KB per code tile
#define BF_DWORDS (NTILES_K * KT_DW)       // 524288 dw = 2 MB

#define AS1 __attribute__((address_space(1)))
#define AS3 __attribute__((address_space(3)))

#if __has_builtin(__builtin_amdgcn_global_load_async_to_lds_b128)
#define VQ_ASYNC_LDS 1
#else
#define VQ_ASYNC_LDS 0
#endif

__device__ __forceinline__ void vq_wait_async() {
#if __has_builtin(__builtin_amdgcn_s_wait_asynccnt)
  __builtin_amdgcn_s_wait_asynccnt(0);
#else
  asm volatile("s_wait_asynccnt 0x0" ::: "memory");
#endif
}

// round-to-nearest-even f32 -> bf16, packed pair {f0 -> [15:0], f1 -> [31:16]}
__device__ __forceinline__ unsigned int packbf(float f0, float f1) {
  unsigned int u0 = __float_as_uint(f0);
  unsigned int u1 = __float_as_uint(f1);
  u0 = (u0 + 0x7FFFu + ((u0 >> 16) & 1u)) >> 16;
  u1 = (u1 + 0x7FFFu + ((u1 >> 16) & 1u)) >> 16;
  return (u0 & 0xFFFFu) | (u1 << 16);
}

// ---------------------------------------------------------------------------
// Prep: swizzle embed into bf16 B-operand register images (ISA 7.12.2 layout).
// ---------------------------------------------------------------------------
__global__ __launch_bounds__(256)
void vq_prep_frag(const float* __restrict__ embed, unsigned int* __restrict__ Bfrag) {
  const int tid = blockIdx.x * 256 + threadIdx.x;
  if (tid >= BF_DWORDS) return;
  const int j    = tid & 7;
  const int lane = (tid >> 3) & 31;
  const int frag = tid >> 8;
  const int kt   = frag >> 4;
  const int c    = frag & 15;
  const int code = kt * 16 + (lane & 15);
  const int half = lane >> 4;
  const int d    = c * 32 + ((j & 4) ? 16 : 0) + half * 8 + (j & 3) * 2;
  const float f0 = embed[(size_t)code * D_ + d];
  const float f1 = embed[(size_t)code * D_ + d + 1];
  Bfrag[tid] = packbf(f0, f1);
}

__global__ __launch_bounds__(256)
void vq_prep_e2(const float* __restrict__ embed, float* __restrict__ e2) {
  const int k = blockIdx.x * 256 + threadIdx.x;
  if (k >= K_) return;
  const float* row = embed + (size_t)k * D_;
  float s = 0.f;
  for (int d = 0; d < D_; ++d) s += row[d] * row[d];
  e2[k] = s;
}

// ---------------------------------------------------------------------------
// Main GEMM + argmax kernel.
// ---------------------------------------------------------------------------
__global__ __launch_bounds__(256)
void vq_gemm_argmax(const float* __restrict__ x,
                    const unsigned int* __restrict__ Bfrag,
                    const float* __restrict__ e2,
                    int* __restrict__ ind) {
  __shared__ unsigned int sL[16384];          // 64 KB; staging then B ring
  const int tid  = threadIdx.x;
  const int lane = tid & 31;
  const int wave = tid >> 5;
  const int m0   = blockIdx.x * 128 + wave * 16;   // global row base (b*N + n)
  const int b    = m0 >> 13;                       // / N_
  const int n0   = m0 & (N_ - 1);
  const float* xb = x + (size_t)b * (size_t)(D_ * N_) + n0;

  const int n    = lane & 15;
  const int half = lane >> 4;

  // ---- Phase 1: build the register-resident A operand (16x512 bf16). -----
  // Two half-D phases so x staging only needs 8 KB/wave (64 KB total).
  v16bf A[NCHUNK_D];
  unsigned int* my = sL + wave * 2048;
  for (int p = 0; p < 2; ++p) {
    const int dp0  = lane >> 2;          // 8 local d-pairs per iteration
    const int noff = (lane & 3) * 4;     // 4 lanes cover the 16 columns
    for (int it = 0; it < 16; ++it) {
      const int dpl = it * 8 + dp0;      // local d-pair 0..127
      const int d   = p * 256 + dpl * 2;
      const float4 a0 = *(const float4*)(xb + (size_t)d * N_ + noff);
      const float4 a1 = *(const float4*)(xb + (size_t)(d + 1) * N_ + noff);
      my[dpl * 16 + noff + 0] = packbf(a0.x, a1.x);
      my[dpl * 16 + noff + 1] = packbf(a0.y, a1.y);
      my[dpl * 16 + noff + 2] = packbf(a0.z, a1.z);
      my[dpl * 16 + noff + 3] = packbf(a0.w, a1.w);
    }
    __syncthreads();
#pragma unroll
    for (int ch = 0; ch < 8; ++ch) {
      v8i w;
#pragma unroll
      for (int j = 0; j < 8; ++j) {
        const int dl = ch * 32 + ((j & 4) ? 16 : 0) + half * 8 + (j & 3) * 2;
        w[j] = (int)my[(dl >> 1) * 16 + n];
      }
      A[p * 8 + ch] = __builtin_bit_cast(v16bf, w);
    }
    __syncthreads();   // staging region is reused (next phase / B ring)
  }

  // ---- Phase 2: K loop over code tiles with double-buffered B ring. ------
  int4 pend[4];  // fallback staging registers (dead in async path)
  auto stage_issue = [&](int kt, int slot) {
    const unsigned int* src = Bfrag + (size_t)kt * KT_DW + tid * 4;
#if VQ_ASYNC_LDS
    unsigned int* dst = sL + slot * 4096 + tid * 4;
#pragma unroll
    for (int r = 0; r < 4; ++r)
      __builtin_amdgcn_global_load_async_to_lds_b128(
          (AS1 vsi4*)(src + r * 1024),
          (AS3 vsi4*)(dst + r * 1024), 0, 0);
#else
#pragma unroll
    for (int r = 0; r < 4; ++r) pend[r] = *(const int4*)(src + r * 1024);
#endif
  };
  auto stage_commit = [&](int slot) {
#if VQ_ASYNC_LDS
    (void)pend;
    vq_wait_async();
#else
    unsigned int* dst = sL + slot * 4096 + tid * 4;
#pragma unroll
    for (int r = 0; r < 4; ++r) *(int4*)(dst + r * 1024) = pend[r];
#endif
  };

  float best[8];
  int   bidx[8];
#pragma unroll
  for (int v = 0; v < 8; ++v) { best[v] = -3.4e38f; bidx[v] = 0; }

  stage_issue(0, 0);
  stage_commit(0);
  __syncthreads();

  for (int kt = 0; kt < NTILES_K; ++kt) {
    const int slot = kt & 1;
    if (kt + 1 < NTILES_K) stage_issue(kt + 1, slot ^ 1);  // prefetch next tile

    const unsigned int* ring = sL + slot * 4096 + lane * 8;
    v8f C = {0.f, 0.f, 0.f, 0.f, 0.f, 0.f, 0.f, 0.f};
#pragma unroll
    for (int c = 0; c < NCHUNK_D; ++c) {
      const v8i wb = *(const v8i*)(ring + c * FRAG_DW);
      C = __builtin_amdgcn_wmma_f32_16x16x32_bf16(
              false, A[c], false, __builtin_bit_cast(v16bf, wb),
              (short)0, C, false, false);
    }

    // score = 2*dot - |e|^2 ; |x|^2 is row-constant, irrelevant for argmax
    const int   code = kt * 16 + n;
    const float e2v  = e2[code];
#pragma unroll
    for (int v = 0; v < 8; ++v) {
      const float s = 2.0f * C[v] - e2v;
      if (s > best[v]) { best[v] = s; bidx[v] = code; }  // strict > keeps lowest code
    }

    if (kt + 1 < NTILES_K) stage_commit(slot ^ 1);
    __syncthreads();
  }

  // Reduce across the 16 lanes sharing each row (min-index tie break).
#pragma unroll
  for (int off = 1; off < 16; off <<= 1) {
#pragma unroll
    for (int v = 0; v < 8; ++v) {
      const float ob = __shfl_xor(best[v], off, 32);
      const int   oi = __shfl_xor(bidx[v], off, 32);
      if (ob > best[v] || (ob == best[v] && oi < bidx[v])) { best[v] = ob; bidx[v] = oi; }
    }
  }
  if (n == 0) {               // lanes 0 and 16 hold rows 0..7 / 8..15
#pragma unroll
    for (int v = 0; v < 8; ++v) ind[m0 + half * 8 + v] = bidx[v];
  }
}

// ---------------------------------------------------------------------------
// Gather: quantize[b][d][n] = embed[ind[b][n]][d]; also emit indices as float.
// ---------------------------------------------------------------------------
__global__ __launch_bounds__(256)
void vq_gather(const float* __restrict__ embed, const int* __restrict__ ind,
               float* __restrict__ outq, float* __restrict__ outind) {
  const long long tid = (long long)blockIdx.x * 256 + threadIdx.x;
  const int nn = (int)(tid & (N_ - 1));
  const int dd = (int)((tid >> 13) & (D_ - 1));   // N_=2^13, D_=2^9
  const int bb = (int)(tid >> 22);
  const int idx = ind[bb * N_ + nn];
  outq[tid] = embed[(size_t)idx * D_ + dd];
  if (tid < (long long)B_ * N_) outind[tid] = (float)ind[tid];
}

// ---------------------------------------------------------------------------
extern "C" void kernel_launch(void* const* d_in, const int* in_sizes, int n_in,
                              void* d_out, int out_size, void* d_ws, size_t ws_size,
                              hipStream_t stream) {
  const float* x     = (const float*)d_in[0];
  const float* embed = (const float*)d_in[1];

  unsigned char* ws = (unsigned char*)d_ws;
  unsigned int* Bfrag = (unsigned int*)ws;                              // 2 MB
  float* e2 = (float*)(ws + (size_t)BF_DWORDS * 4);                     // 8 KB
  int*   ind = (int*)(ws + (size_t)BF_DWORDS * 4 + (size_t)K_ * 4);     // 512 KB

  float* outq   = (float*)d_out;
  float* outind = outq + (size_t)B_ * D_ * N_;

  vq_prep_frag<<<(BF_DWORDS + 255) / 256, 256, 0, stream>>>(embed, Bfrag);
  vq_prep_e2<<<(K_ + 255) / 256, 256, 0, stream>>>(embed, e2);
  vq_gemm_argmax<<<(B_ * N_) / 128, 256, 0, stream>>>(x, Bfrag, e2, ind);
  vq_gather<<<(int)(((size_t)B_ * D_ * N_) / 256), 256, 0, stream>>>(embed, ind, outq, outind);
}
